// LaneLoss_80839874445934
// MI455X (gfx1250) — matrix operand order repaired and checked by reading, not compiled, not run on hardware
//
#include <hip/hip_runtime.h>

typedef __attribute__((ext_vector_type(16))) _Float16 v16h;
typedef __attribute__((ext_vector_type(8)))  float    v8f;

#define Q_N 100
#define G_N 50
#define P_N 24
#define RS_N 96
#define GH 72
#define GW 128
#define KDIM (GH*GW)      // 9216
#define QPAD 112          // 7 * 16
#define GPAD 64           // 4 * 16

// ---------------------------------------------------------------------------
// Kernel 1: prepare gt mask (with <2-visible fallback) + arc-length resample
// blocks: 150 (100 pred + 50 gt), 96 threads
// ---------------------------------------------------------------------------
__global__ void lane_resample_kernel(const float* __restrict__ pred_pts,
                                     const float* __restrict__ gt_pts,
                                     const float* __restrict__ gt_vis,
                                     float* __restrict__ predRs,
                                     float* __restrict__ gtRs,
                                     float* __restrict__ mGt) {
  const int b = blockIdx.x;
  const bool isPred = (b < Q_N);
  const int gi = b - Q_N;
  const float* src = isPred ? (pred_pts + b * (P_N * 2)) : (gt_pts + gi * (P_N * 2));
  __shared__ float sp[P_N * 2];
  __shared__ float sm[P_N];
  __shared__ float cum[P_N];
  const int t = threadIdx.x;

  if (t < P_N * 2) sp[t] = src[t];
  if (t < P_N) {
    sm[t] = isPred ? 1.0f : ((gt_vis[gi * P_N + t] > 0.5f) ? 1.0f : 0.0f);
  }
  __syncthreads();

  if (t == 0 && !isPred) {
    float cnt = 0.f;
    for (int i = 0; i < P_N; ++i) cnt += sm[i];
    if (cnt < 2.0f)
      for (int i = 0; i < P_N; ++i) sm[i] = 1.0f;
  }
  __syncthreads();

  if (!isPred && t < P_N) mGt[gi * P_N + t] = sm[t];
  if (t == 0) {
    cum[0] = 0.0f;
    for (int i = 0; i < P_N - 1; ++i) {
      float dx = sp[2 * i + 2] - sp[2 * i];
      float dy = sp[2 * i + 3] - sp[2 * i + 1];
      float len = sqrtf(fmaxf(dx * dx + dy * dy, 1e-12f)) * (sm[i] * sm[i + 1]);
      cum[i + 1] = cum[i] + len;
    }
  }
  __syncthreads();

  const float total = cum[P_N - 1];
  const float tt = ((float)t * (1.0f / (RS_N - 1))) * total;
  int idx = 0;
  for (int i = 1; i < P_N; ++i)
    if (cum[i] <= tt) idx = i;          // searchsorted(side='right') - 1
  if (idx > P_N - 2) idx = P_N - 2;

  float lx = sp[2 * idx],     ly = sp[2 * idx + 1];
  float rx = sp[2 * idx + 2], ry = sp[2 * idx + 3];
  float lt = cum[idx], rt = cum[idx + 1];
  float alpha = (tt - lt) / fmaxf(rt - lt, 1e-8f);
  float ox = lx + alpha * (rx - lx);
  float oy = ly + alpha * (ry - ly);
  if (total < 1e-8f) { ox = sp[0]; oy = sp[1]; }

  float* dst = isPred ? (predRs + b * (RS_N * 2)) : (gtRs + gi * (RS_N * 2));
  dst[2 * t] = ox;
  dst[2 * t + 1] = oy;
}

// ---------------------------------------------------------------------------
// Kernel 2: soft lane mask over the 72x128 grid, stored as f16 row-major,
// padded rows zeroed. grid = (KDIM/256, nRowsPad), 256 threads.
// ---------------------------------------------------------------------------
__global__ void lane_softmask_kernel(const float* __restrict__ pts,
                                     const float* __restrict__ mask, // null => all-ones
                                     int nLanes,
                                     _Float16* __restrict__ out) {
  const int row = blockIdx.y;
  const int k = blockIdx.x * blockDim.x + threadIdx.x; // < 9216 exactly
  if (row >= nLanes) { out[(size_t)row * KDIM + k] = (_Float16)0.0f; return; }

  __shared__ float sp[P_N * 2];
  __shared__ float sm[P_N];
  const int t = threadIdx.x;
  if (t < P_N * 2) sp[t] = pts[row * (P_N * 2) + t];
  if (t < P_N) sm[t] = mask ? mask[row * P_N + t] : 1.0f;
  __syncthreads();

  const float px = (float)(k % GW) * (1.0f / (GW - 1));
  const float py = (float)(k / GW) * (1.0f / (GH - 1));

  float minSeg = 3.0e38f, minPt = 3.0e38f;
  bool anySeg = false;
  for (int i = 0; i < P_N - 1; ++i) {
    if (sm[i] * sm[i + 1] > 0.5f) {
      anySeg = true;
      float ax = sp[2 * i], ay = sp[2 * i + 1];
      float abx = sp[2 * i + 2] - ax, aby = sp[2 * i + 3] - ay;
      float denom = fmaxf(abx * abx + aby * aby, 1e-8f);
      float tp = fminf(fmaxf(((px - ax) * abx + (py - ay) * aby) / denom, 0.f), 1.f);
      float dx = px - (ax + tp * abx), dy = py - (ay + tp * aby);
      minSeg = fminf(minSeg, sqrtf(fmaxf(dx * dx + dy * dy, 1e-12f)));
    }
  }
  for (int i = 0; i < P_N; ++i) {
    if (sm[i] > 0.5f) {
      float dx = px - sp[2 * i], dy = py - sp[2 * i + 1];
      minPt = fminf(minPt, sqrtf(fmaxf(dx * dx + dy * dy, 1e-12f)));
    }
  }
  const float mind = anySeg ? minSeg : minPt;
  const float val = 1.0f / (1.0f + expf(-(0.03f - mind) * 80.0f));
  out[(size_t)row * KDIM + k] = (_Float16)val;
}

// ---------------------------------------------------------------------------
// Kernel 3: deterministic per-row sum of the f16 mask (union term).
// ---------------------------------------------------------------------------
__global__ void lane_rowsum_kernel(const _Float16* __restrict__ m,
                                   float* __restrict__ sums) {
  const int row = blockIdx.x;
  __shared__ float red[256];
  float s = 0.f;
  for (int k = threadIdx.x; k < KDIM; k += 256)
    s += (float)m[(size_t)row * KDIM + k];
  red[threadIdx.x] = s;
  __syncthreads();
  for (int off = 128; off > 0; off >>= 1) {
    if (threadIdx.x < off) red[threadIdx.x] += red[threadIdx.x + off];
    __syncthreads();
  }
  if (threadIdx.x == 0) sums[row] = red[0];
}

// ---------------------------------------------------------------------------
// Kernel 4: inter = pm @ gm^T via v_wmma_f32_16x16x32_f16.
// grid = (4, 7), one wave (32 threads) per 16x16 output tile, K = 9216.
// Fragment packing per CDNA5 ISA 16-bit A(16x32)/B(32x16) layouts.
// ---------------------------------------------------------------------------
union FragH {
  v16h h;
  unsigned u[8];
  uint4 q[2];
};

__global__ void lane_wmma_gemm_kernel(const _Float16* __restrict__ pm,
                                      const _Float16* __restrict__ gm,
                                      float* __restrict__ inter) {
  const int tn = blockIdx.x;            // 0..3  (gt cols)
  const int tm = blockIdx.y;            // 0..6  (pred rows)
  const int l = threadIdx.x;            // 0..31
  const int hl = l >> 4;                // lane half
  const int l16 = l & 15;

  const _Float16* aRow = pm + (size_t)(tm * 16 + l16) * KDIM;
  const _Float16* bRow = gm + (size_t)(tn * 16 + l16) * KDIM;

  v8f c = {};
  for (int k0 = 0; k0 < KDIM; k0 += 32) {
    FragH a, b;
#pragma unroll
    for (int v = 0; v < 8; ++v) {
      // A 16x32 f16 layout: K = 16*(v/4) + 8*(lane/16) + 2*(v%4) + half
      const int ks = ((v & 4) ? 16 : 0) + (hl ? 8 : 0) + 2 * (v & 3);
      a.u[v] = *reinterpret_cast<const unsigned*>(aRow + k0 + ks);
    }
    // B 32x16 f16 layout: lane holds N=l%16, K = 16*(e/8) + 8*(lane/16) + e%8
    const uint4* bp = reinterpret_cast<const uint4*>(bRow + k0 + (hl ? 8 : 0));
    b.q[0] = bp[0];
    b.q[1] = bp[2];
    c = __builtin_amdgcn_wmma_f32_16x16x32_f16(false, a.h, false, b.h,
                                               (short)0, c, false, false);
  }
#pragma unroll
  for (int v = 0; v < 8; ++v) {
    const int m = (hl ? 8 : 0) + v;     // C/D layout: VGPR v -> M = v + 8*(lane/16)
    inter[(size_t)(tm * 16 + m) * GPAD + tn * 16 + l16] = c[v];
  }
}

// ---------------------------------------------------------------------------
// Kernel 5: per-pair sym/tan/curv + overlap -> final cost.
// grid = (G, Q), 96 threads (one per resample point).
// ---------------------------------------------------------------------------
__device__ __forceinline__ float lane_min_dist_poly(float px, float py,
                                                    const float* __restrict__ P) {
  float md = 3.0e38f;
  for (int i = 0; i < RS_N - 1; ++i) {
    float ax = P[2 * i], ay = P[2 * i + 1];
    float abx = P[2 * i + 2] - ax, aby = P[2 * i + 3] - ay;
    float denom = fmaxf(abx * abx + aby * aby, 1e-8f);
    float tp = fminf(fmaxf(((px - ax) * abx + (py - ay) * aby) / denom, 0.f), 1.f);
    float dx = px - (ax + tp * abx), dy = py - (ay + tp * aby);
    md = fminf(md, sqrtf(fmaxf(dx * dx + dy * dy, 1e-12f)));
  }
  return md;
}

__device__ __forceinline__ float lane_smooth_l1(float x) {
  float a = fabsf(x);
  return (a < 1.0f) ? 0.5f * a * a : a - 0.5f;
}

__global__ void lane_combine_kernel(const float* __restrict__ predRs,
                                    const float* __restrict__ gtRs,
                                    const float* __restrict__ inter,
                                    const float* __restrict__ pmSum,
                                    const float* __restrict__ gmSum,
                                    float* __restrict__ out) {
  const int g = blockIdx.x;
  const int q = blockIdx.y;
  const int t = threadIdx.x;
  __shared__ float A[RS_N * 2], B[RS_N * 2];
  __shared__ float rA[RS_N], rB[RS_N], rT[RS_N], rC[RS_N];

  A[2 * t]     = predRs[q * (RS_N * 2) + 2 * t];
  A[2 * t + 1] = predRs[q * (RS_N * 2) + 2 * t + 1];
  B[2 * t]     = gtRs[g * (RS_N * 2) + 2 * t];
  B[2 * t + 1] = gtRs[g * (RS_N * 2) + 2 * t + 1];
  __syncthreads();

  const float dA = lane_min_dist_poly(A[2 * t], A[2 * t + 1], B);
  const float dB = lane_min_dist_poly(B[2 * t], B[2 * t + 1], A);

  const int ip = (t + 1 < RS_N) ? t + 1 : RS_N - 1;
  const int im = (t - 1 > 0) ? t - 1 : 0;
  float tax = A[2 * ip] - A[2 * im], tay = A[2 * ip + 1] - A[2 * im + 1];
  float na = fmaxf(sqrtf(tax * tax + tay * tay), 1e-8f);
  float tbx = B[2 * ip] - B[2 * im], tby = B[2 * ip + 1] - B[2 * im + 1];
  float nb = fmaxf(sqrtf(tbx * tbx + tby * tby), 1e-8f);
  const float td = fabsf((tax * tbx + tay * tby) / (na * nb));

  float cv = 0.f;
  if (t < RS_N - 2) {
    float psx = A[2 * (t + 2)]     - 2.f * A[2 * (t + 1)]     + A[2 * t];
    float psy = A[2 * (t + 2) + 1] - 2.f * A[2 * (t + 1) + 1] + A[2 * t + 1];
    float gsx = B[2 * (t + 2)]     - 2.f * B[2 * (t + 1)]     + B[2 * t];
    float gsy = B[2 * (t + 2) + 1] - 2.f * B[2 * (t + 1) + 1] + B[2 * t + 1];
    cv = lane_smooth_l1(psx - gsx) + lane_smooth_l1(psy - gsy);
  }
  rA[t] = dA; rB[t] = dB; rT[t] = td; rC[t] = cv;
  __syncthreads();

  if (t == 0) {
    float sA = 0, sB = 0, sT = 0, sC = 0;
    for (int i = 0; i < RS_N; ++i) { sA += rA[i]; sB += rB[i]; sT += rT[i]; sC += rC[i]; }
    const float sym  = 0.5f * (sA * (1.0f / RS_N) + sB * (1.0f / RS_N));
    const float tanc = 1.0f - sT * (1.0f / RS_N);
    const float curv = sC * (1.0f / ((RS_N - 2) * 2));
    const float I = inter[(size_t)q * GPAD + g];
    const float U = pmSum[q] + gmSum[g] - I;
    const float ovl = 1.0f - I / fmaxf(U, 1e-8f);
    out[q * G_N + g] = 5.0f * sym + 1.0f * tanc + 0.5f * curv + 2.0f * ovl;
  }
}

// ---------------------------------------------------------------------------
extern "C" void kernel_launch(void* const* d_in, const int* in_sizes, int n_in,
                              void* d_out, int out_size, void* d_ws, size_t ws_size,
                              hipStream_t stream) {
  const float* pred_pts = (const float*)d_in[0]; // (100,24,2)
  const float* gt_pts   = (const float*)d_in[1]; // (50,24,2)
  const float* gt_vis   = (const float*)d_in[2]; // (50,24)
  float* out = (float*)d_out;                    // (100,50)

  char* ws = (char*)d_ws;
  size_t off = 0;
  auto alloc = [&](size_t bytes) -> char* {
    char* p = ws + off;
    off = (off + bytes + 255) & ~(size_t)255;
    return p;
  };
  float*     predRs = (float*)alloc(Q_N * RS_N * 2 * sizeof(float));
  float*     gtRs   = (float*)alloc(G_N * RS_N * 2 * sizeof(float));
  float*     mGt    = (float*)alloc(G_N * P_N * sizeof(float));
  float*     pmSum  = (float*)alloc(Q_N * sizeof(float));
  float*     gmSum  = (float*)alloc(G_N * sizeof(float));
  float*     inter  = (float*)alloc((size_t)QPAD * GPAD * sizeof(float));
  _Float16*  pmF16  = (_Float16*)alloc((size_t)QPAD * KDIM * sizeof(_Float16));
  _Float16*  gmF16  = (_Float16*)alloc((size_t)GPAD * KDIM * sizeof(_Float16));

  // 1) resample pred+gt polylines (and gt visibility mask with fallback)
  lane_resample_kernel<<<Q_N + G_N, RS_N, 0, stream>>>(pred_pts, gt_pts, gt_vis,
                                                       predRs, gtRs, mGt);
  // 2) soft masks as f16 (rows padded with zeros for WMMA tiling)
  lane_softmask_kernel<<<dim3(KDIM / 256, QPAD), 256, 0, stream>>>(
      pred_pts, nullptr, Q_N, pmF16);
  lane_softmask_kernel<<<dim3(KDIM / 256, GPAD), 256, 0, stream>>>(
      gt_pts, mGt, G_N, gmF16);
  // 3) row sums for the IoU union
  lane_rowsum_kernel<<<Q_N, 256, 0, stream>>>(pmF16, pmSum);
  lane_rowsum_kernel<<<G_N, 256, 0, stream>>>(gmF16, gmSum);
  // 4) inter = pm @ gm^T via WMMA
  lane_wmma_gemm_kernel<<<dim3(GPAD / 16, QPAD / 16), 32, 0, stream>>>(
      pmF16, gmF16, inter);
  // 5) final per-pair cost
  lane_combine_kernel<<<dim3(G_N, Q_N), RS_N, 0, stream>>>(
      predRs, gtRs, inter, pmSum, gmSum, out);
}
